// CouncilLayer_17798344475058
// MI455X (gfx1250) — compile-verified, matrix operand order checked by reading, not run on hardware
//
#include <hip/hip_runtime.h>
#include <stdint.h>

// ---------------------------------------------------------------------------
// CouncilLayer (dense all-expert MoE FFN) for MI455X / gfx1250
//   out[n,c] = sum_e g[n,e] * ( gelu(x[n]@w1[e] + b1[e]) @ w2[e] + b2[e] )[c]
// N=2048 tokens, C=768, E=32, H=3072.  ~620 GFLOP, ~1000 FLOP/byte =>
// compute bound => bf16 WMMA (16x16x32, f32 accum).
// Weights are converted fp32->bf16 AND transposed once per call so the WMMA
// B-matrix K dimension is memory-contiguous: every A/B fragment is exactly
// 2x ds_load_b128.  Weight tiles are staged global->LDS by the Tensor Data
// Mover in a flat 12-block software pipeline (6 w1 + 6 w2 tiles per H-chunk,
// next chunk's first tile prefetched during the last block) so every DMA
// overlaps the WMMA stream.
// ---------------------------------------------------------------------------

#define HAVE_TDM 0
#if defined(__has_builtin)
#  if __has_builtin(__builtin_amdgcn_tensor_load_to_lds) && __has_builtin(__builtin_amdgcn_s_wait_tensorcnt)
#    undef HAVE_TDM
#    define HAVE_TDM 1
#  endif
#endif

typedef __bf16 bf16_t;
typedef bf16_t v16bf __attribute__((ext_vector_type(16)));
typedef float  v8f   __attribute__((ext_vector_type(8)));
typedef unsigned int u32x4 __attribute__((ext_vector_type(4)));
typedef int          i32x4 __attribute__((ext_vector_type(4)));
typedef int          i32x8 __attribute__((ext_vector_type(8)));

union Frag {
  v16bf v;
  unsigned int d[8];
};

#define NTOK 2048
#define CC   768
#define EE   32
#define HH   3072

#define MT   32      // tokens per workgroup tile
#define HC   128     // H chunk width
#define KB   128     // K block staged in LDS
#define NHC  (HH / HC)   // 24 H chunks
#define WST  136     // lds_w row stride in halves (128 + 8 pad -> 4-bank skew)
#define XST  776     // x_sh row stride in halves (768 + 8 pad)
#define HST  136     // h_sh row stride in halves

// fp32 -> bf16 round-to-nearest-even
__device__ __forceinline__ unsigned short f2bf(float x) {
  unsigned u = __float_as_uint(x);
  unsigned r = u + 0x7FFFu + ((u >> 16) & 1u);
  return (unsigned short)(r >> 16);
}

// 16-bit WMMA fragment (ISA 7.12.2): lane (nl,al) holds halves
//   d[0..3] = K[8a .. 8a+7], d[4..7] = K[16+8a .. 16+8a+7] of row `row`.
// With K contiguous in LDS this is two aligned b128 loads.
__device__ __forceinline__ void load_frag(Frag& f, const unsigned short* lds,
                                          int row, int k0, int stride, int a) {
  const int base = row * stride + k0 + 8 * a;   // 8-half (16B) aligned
  const uint4 lo = *(const uint4*)(lds + base);
  const uint4 hi = *(const uint4*)(lds + base + 16);
  f.d[0] = lo.x; f.d[1] = lo.y; f.d[2] = lo.z; f.d[3] = lo.w;
  f.d[4] = hi.x; f.d[5] = hi.y; f.d[6] = hi.z; f.d[7] = hi.w;
}

#if HAVE_TDM
// Tensor Data Mover: 2D bf16 tile [tile1 rows x 128 elems], global -> LDS,
// LDS padding: pad_interval code 5 (64 DW = one 128-half row), pad_amount
// code 3 (4 DW = 8 halves) => padded LDS row stride = 136 halves (WST).
__device__ __forceinline__ void tdm_load_2d(unsigned lds_addr, const void* gp,
                                            unsigned tile0, unsigned tile1,
                                            unsigned long long stride0) {
  const unsigned long long ga = (unsigned long long)gp;
  const unsigned tdim0 = (unsigned)stride0;   // >= tile0, no OOB
  const unsigned tdim1 = tile1;
  u32x4 g0;
  g0[0] = 1u;                                  // count=1, user descriptor
  g0[1] = lds_addr;                            // LDS byte address
  g0[2] = (unsigned)ga;                        // global addr [31:0]
  g0[3] = (unsigned)((ga >> 32) & 0x1FFFFFFull) | (2u << 30);  // addr[56:32] | type=2
  i32x8 g1;
  g1[0] = (int)((1u << 16) |                   // data_size = 1 (2 bytes)
                (1u << 20) |                   // pad_enable
                (5u << 22) |                   // pad_interval: 64 DWORDs
                (3u << 25));                   // pad_amount: 4 DWORDs
  g1[1] = (int)((tdim0 & 0xFFFFu) << 16);      // tensor_dim0[15:0]
  g1[2] = (int)((tdim0 >> 16) | ((tdim1 & 0xFFFFu) << 16));
  g1[3] = (int)(((tdim1 >> 16) & 0xFFFFu) | (tile0 << 16));
  g1[4] = (int)(tile1 & 0xFFFFu);              // tile_dim1 (tile_dim2 = 0)
  g1[5] = (int)(stride0 & 0xFFFFFFFFull);      // tensor_dim0_stride[31:0]
  g1[6] = (int)((stride0 >> 32) & 0xFFFFull);
  g1[7] = 0;
  i32x4 z4 = {0, 0, 0, 0};
  i32x8 z8 = {0, 0, 0, 0, 0, 0, 0, 0};
  __builtin_amdgcn_tensor_load_to_lds(g0, g1, z4, z4, z8, 0);
}
#endif

// Issue staging of a [128 x 128] bf16 block into `lds` (padded stride WST).
// TDM path: wave 0 fires one DMA descriptor (completion via s_wait_tensorcnt).
__device__ __forceinline__ void stage_issue(unsigned short* lds,
                                            const unsigned short* g,
                                            unsigned long long row_stride,
                                            int tid, int wave) {
#if HAVE_TDM
  if (wave == 0)
    tdm_load_2d((unsigned)(uintptr_t)lds, (const void*)g, KB, KB, row_stride);
#else
  for (int idx = tid; idx < KB * 16; idx += 256) {
    const int row = idx >> 4, c8 = (idx & 15) * 8;
    *(uint4*)&lds[row * WST + c8] =
        *(const uint4*)&g[(size_t)row * row_stride + c8];
  }
#endif
}

__device__ __forceinline__ void stage_wait(int wave) {
#if HAVE_TDM
  if (wave == 0) __builtin_amdgcn_s_wait_tensorcnt(0);
#endif
}

// ---------------------------------------------------------------------------
// Kernel 1: fused fp32->bf16 convert + transpose (tiled, coalesced both ways)
// src: [E][R][S] fp32  ->  dst: [E][S][R] bf16
// ---------------------------------------------------------------------------
__global__ void transpose_bf16_kernel(const float* __restrict__ src,
                                      unsigned short* __restrict__ dst,
                                      int R, int S) {
  __shared__ float tile[32][33];
  const int e = blockIdx.z;
  const int s0 = blockIdx.x * 32, r0 = blockIdx.y * 32;
  const int tx = threadIdx.x, ty = threadIdx.y;   // 32 x 8
  const float* se = src + (size_t)e * R * S;
  unsigned short* de = dst + (size_t)e * R * S;
#pragma unroll
  for (int k = 0; k < 4; ++k)
    tile[ty + 8 * k][tx] = se[(size_t)(r0 + ty + 8 * k) * S + s0 + tx];
  __syncthreads();
#pragma unroll
  for (int k = 0; k < 4; ++k)
    de[(size_t)(s0 + ty + 8 * k) * R + r0 + tx] = f2bf(tile[tx][ty + 8 * k]);
}

// ---------------------------------------------------------------------------
// Kernel 2: gating softmax (fp32), x -> bf16, out := sum_e g_e * b2[e,:]
// One wave32 per token.
// ---------------------------------------------------------------------------
__global__ void gate_kernel(const float* __restrict__ x,
                            const float* __restrict__ gw,
                            const float* __restrict__ gb,
                            const float* __restrict__ b2,
                            float* __restrict__ out,
                            unsigned short* __restrict__ xbf,
                            float* __restrict__ gates) {
  const int n = blockIdx.x;
  const int lane = threadIdx.x;                 // 0..31, lane == expert
  const float* xr = x + (size_t)n * CC;

  for (int c = lane; c < CC; c += 32) xbf[(size_t)n * CC + c] = f2bf(xr[c]);

  float acc = gb[lane];
  for (int c = 0; c < CC; ++c) acc += xr[c] * gw[c * EE + lane];

  float m = acc;
  for (int o = 16; o > 0; o >>= 1) m = fmaxf(m, __shfl_xor(m, o, 32));
  const float p = expf(acc - m);
  float s = p;
  for (int o = 16; o > 0; o >>= 1) s += __shfl_xor(s, o, 32);
  const float g = p / s;
  gates[(size_t)n * EE + lane] = g;

  // out init = gate-weighted b2 (folds the second bias out of the GEMM kernel)
  float a2[24];
#pragma unroll
  for (int i = 0; i < 24; ++i) a2[i] = 0.f;
  for (int e = 0; e < EE; ++e) {
    const float ge = __shfl(g, e, 32);
    const float* b2e = b2 + (size_t)e * CC;
#pragma unroll
    for (int i = 0; i < 24; ++i) a2[i] += ge * b2e[lane + 32 * i];
  }
#pragma unroll
  for (int i = 0; i < 24; ++i) out[(size_t)n * CC + lane + 32 * i] = a2[i];
}

// ---------------------------------------------------------------------------
// Kernel 3: fused per-expert FFN with bf16 WMMA.
// grid = (64 token tiles, 32 experts), block = 256 threads = 8 wave32s.
// Wave (r,q): r = 16-token row band, q = column group.
// w1t: [E][H][C] bf16 (K=C contiguous), w2t: [E][C][H] bf16 (K=H contiguous).
// Per H-chunk: flat 12-block pipeline, block i+1's TDM in flight during
// block i's 8 WMMAs; next chunk's block 0 prefetched during block 11.
// ---------------------------------------------------------------------------
__global__ __launch_bounds__(256) void moe_kernel(
    const unsigned short* __restrict__ xbf,
    const unsigned short* __restrict__ w1t,
    const unsigned short* __restrict__ w2t,
    const float* __restrict__ b1,
    const float* __restrict__ gates,
    float* __restrict__ out) {
  __shared__ __attribute__((aligned(16))) unsigned short lds_w[2][KB * WST];
  __shared__ __attribute__((aligned(16))) unsigned short x_sh[MT * XST];
  __shared__ __attribute__((aligned(16))) unsigned short h_sh[MT * HST];

  const int tid  = threadIdx.x;
  const int lane = tid & 31, wave = tid >> 5;
  const int r = wave >> 2, q = wave & 3;        // 2 row bands x 4 col groups
  const int nl = lane & 15, al = lane >> 4;
  const int e    = blockIdx.y;
  const int tok0 = blockIdx.x * MT;

  const unsigned short* w1e = w1t + (size_t)e * HH * CC;
  const unsigned short* w2e = w2t + (size_t)e * CC * HH;

  // pipeline prologue: first w1 block of chunk 0 (only non-overlapped DMA)
  stage_issue(lds_w[0], w1e, CC, tid, wave);

  // stage x tile [MT x 768] -> LDS with padded stride (conflict-free gathers)
  for (int c8 = tid; c8 < MT * 96; c8 += 256) {
    const int row = c8 / 96, col = (c8 % 96) * 8;
    *(uint4*)&x_sh[row * XST + col] =
        *(const uint4*)(xbf + (size_t)(tok0 + row) * CC + col);
  }

  // this lane's 8 gate values (one per accumulator row)
  float gv[8];
#pragma unroll
  for (int j = 0; j < 8; ++j)
    gv[j] = gates[(size_t)(tok0 + 16 * r + j + 8 * al) * EE + e];

  const v8f vzero = {0.f, 0.f, 0.f, 0.f, 0.f, 0.f, 0.f, 0.f};
  v8f yacc[6][2];
#pragma unroll
  for (int nb = 0; nb < 6; ++nb) {
    yacc[nb][0] = vzero;
    yacc[nb][1] = vzero;
  }

  stage_wait(wave);
  __syncthreads();          // block 0 ready; x_sh + gv published

  for (int hc = 0; hc < NHC; ++hc) {
    const int h0 = hc * HC;
    v8f hacc[2] = {vzero, vzero};

#pragma unroll
    for (int i = 0; i < 12; ++i) {
      // ---- issue next block's DMA (overlaps this block's WMMAs) ----
      unsigned short* nxtb = lds_w[(i + 1) & 1];
      bool have_next = true;
      if (i < 5)
        stage_issue(nxtb, w1e + (size_t)h0 * CC + (i + 1) * KB, CC, tid, wave);
      else if (i == 5)
        stage_issue(nxtb, w2e + h0, HH, tid, wave);
      else if (i < 11)
        stage_issue(nxtb, w2e + (size_t)((i - 5) * KB) * HH + h0, HH, tid, wave);
      else if (hc + 1 < NHC)
        stage_issue(nxtb, w1e + (size_t)(h0 + HC) * CC, CC, tid, wave);
      else
        have_next = false;

      // ---- compute on current block ----
      const unsigned short* cur = lds_w[i & 1];
#pragma unroll
      for (int k32 = 0; k32 < 4; ++k32) {
        Frag A;
        if (i < 6)
          load_frag(A, x_sh, 16 * r + nl, i * KB + k32 * 32, XST, al);
        else
          load_frag(A, h_sh, 16 * r + nl, k32 * 32, HST, al);
#pragma unroll
        for (int t = 0; t < 2; ++t) {
          Frag B;
          load_frag(B, cur, q * 32 + t * 16 + nl, k32 * 32, WST, al);
          v8f& acc = (i < 6) ? hacc[t] : yacc[i - 6][t];
          acc = __builtin_amdgcn_wmma_f32_16x16x32_bf16(
              false, A.v, false, B.v, (short)0, acc, false, false);
        }
      }

      // ---- after last GEMM1 block: +b1, exact-erf GELU, x gate -> h_sh ----
      // (published by this block's barrier, read from block 6 onward)
      if (i == 5) {
#pragma unroll
        for (int t = 0; t < 2; ++t) {
          const float bb = b1[(size_t)e * HH + h0 + q * 32 + t * 16 + nl];
#pragma unroll
          for (int j = 0; j < 8; ++j) {
            float v = hacc[t][j] + bb;
            v = 0.5f * v * (1.0f + erff(v * 0.70710678118654752f));
            v *= gv[j];
            h_sh[(16 * r + j + 8 * al) * HST + q * 32 + t * 16 + nl] = f2bf(v);
          }
        }
      }

      __syncthreads();                       // all done reading cur / h_sh out
      if (have_next) { stage_wait(wave); __syncthreads(); }  // next buf ready
    }
  }

  // ---- writeback: out += g * y (out pre-initialized with gate-weighted b2)
#pragma unroll
  for (int nb = 0; nb < 6; ++nb)
#pragma unroll
    for (int t = 0; t < 2; ++t)
#pragma unroll
      for (int j = 0; j < 8; ++j) {
        const int col = nb * KB + q * 32 + t * 16 + nl;
        const int tok = tok0 + 16 * r + j + 8 * al;
        atomicAdd(&out[(size_t)tok * CC + col], yacc[nb][t][j]);
      }
}

// ---------------------------------------------------------------------------
extern "C" void kernel_launch(void* const* d_in, const int* in_sizes, int n_in,
                              void* d_out, int out_size, void* d_ws,
                              size_t ws_size, hipStream_t stream) {
  (void)in_sizes; (void)n_in; (void)out_size; (void)ws_size;
  const float* x  = (const float*)d_in[0];
  const float* gw = (const float*)d_in[1];
  const float* gb = (const float*)d_in[2];
  const float* w1 = (const float*)d_in[3];
  const float* b1 = (const float*)d_in[4];
  const float* w2 = (const float*)d_in[5];
  const float* b2 = (const float*)d_in[6];
  float* out = (float*)d_out;

  // workspace carve (all 16B aligned)
  char* ws = (char*)d_ws;
  size_t off = 0;
  unsigned short* xbf = (unsigned short*)(ws + off);
  off += (size_t)NTOK * CC * 2;                       //   3,145,728
  float* gates = (float*)(ws + off);
  off += (size_t)NTOK * EE * 4;                       //     262,144
  unsigned short* w1t = (unsigned short*)(ws + off);  // [E][H][C] bf16
  off += (size_t)EE * CC * HH * 2;                    // 150,994,944
  unsigned short* w2t = (unsigned short*)(ws + off);  // [E][C][H] bf16

  // fp32 -> bf16 with transpose (one pass, replaces plain conversion)
  {
    dim3 b(32, 8);
    dim3 g1(HH / 32, CC / 32, EE);   // w1 [E][C][H] -> w1t [E][H][C]
    transpose_bf16_kernel<<<g1, b, 0, stream>>>(w1, w1t, CC, HH);
    dim3 g2(CC / 32, HH / 32, EE);   // w2 [E][H][C] -> w2t [E][C][H]
    transpose_bf16_kernel<<<g2, b, 0, stream>>>(w2, w2t, HH, CC);
  }

  gate_kernel<<<NTOK, 32, 0, stream>>>(x, gw, gb, b2, out, xbf, gates);

  dim3 grid(NTOK / MT, EE);   // token tiles fast-varying => per-expert L2 reuse
  moe_kernel<<<grid, 256, 0, stream>>>(xbf, w1t, w2t, b1, gates, out);
}
